// CausalSelfAttention_90340342104251
// MI455X (gfx1250) — matrix operand order repaired and checked by reading, not compiled, not run on hardware
//
#include <hip/hip_runtime.h>

typedef __bf16 bf16;
typedef __attribute__((ext_vector_type(16))) bf16  v16bf;
typedef __attribute__((ext_vector_type(8)))  float v8f;

#define B_   4
#define T_   2048
#define D_   1024
#define H_   16
#define HD_  64
#define BT_  (B_ * T_)

// ---- workspace layout (bytes) ----
#define XB_OFF  0ull                                   // x bf16       : 16 MiB
#define WQ_OFF  (XB_OFF + (size_t)BT_ * D_ * 2)        // Wq bf16      :  2 MiB
#define WK_OFF  (WQ_OFF + (size_t)D_ * D_ * 2)
#define WV_OFF  (WK_OFF + (size_t)D_ * D_ * 2)
#define WO_OFF  (WV_OFF + (size_t)D_ * D_ * 2)
#define QB_OFF  (WO_OFF + (size_t)D_ * D_ * 2)         // Q bf16 (b,h,t,d), pre-scaled
#define KB_OFF  (QB_OFF + (size_t)BT_ * D_ * 2)        // K bf16 (b,h,t,d)
#define VT_OFF  (KB_OFF + (size_t)BT_ * D_ * 2)        // V bf16 (b,h,d,t)
#define AB_OFF  (VT_OFF + (size_t)BT_ * D_ * 2)        // attn_out bf16 (b,t,D)

// d_out regions (floats): out | k | v
#define OUTK_ELEM_OFF ((size_t)BT_ * D_)
#define OUTV_ELEM_OFF ((size_t)2 * BT_ * D_)

// LDS A-tile: 32 rows x 32 bf16 (64B) padded to 80B per row -> 40 bf16 stride
#define AT_STRIDE 40

__device__ __forceinline__ v8f wmma_bf16(v16bf a, v16bf b, v8f c) {
  return __builtin_amdgcn_wmma_f32_16x16x32_bf16(false, a, false, b, (short)0, c,
                                                 false, false);
}

// ---- CDNA5 async global->LDS copy (16B per lane), via inline asm ----
// VGLOBAL async encoding: vdst = LDS byte address VGPR, vaddr = 64-bit global
// address (GV mode, saddr = off). Flat form of an LDS pointer on amdgcn is
// {aperture_hi32, lds_offset32}, so truncating the generic pointer to 32 bits
// yields the dsaddr operand.
__device__ __forceinline__ void async_copy16(const bf16* g, bf16* l) {
  const unsigned lds_off = (unsigned)(unsigned long long)l;
  asm volatile("global_load_async_to_lds_b128 %0, %1, off"
               :
               : "v"(lds_off), "v"(g)
               : "memory");
}

__device__ __forceinline__ void wait_async0() {
  asm volatile("s_wait_asynccnt 0x0" ::: "memory");
}

// Bare workgroup split-barrier: orders LDS buffer reuse between waves without
// __syncthreads()'s global-memory fence (which forces loadcnt->0 and would
// serialize the pipelined B-fragment loads). LDS reads are consumed by WMMAs
// (dscnt waits at use) before any wave can reach its next signal, so no
// hardware memory wait is needed at the barrier itself.
__device__ __forceinline__ void block_barrier() {
  asm volatile("s_barrier_signal -1\n\ts_barrier_wait -1" ::: "memory");
}

// A fragment (global): 16x32 bf16, row M = lane&15, half = lane>>4.
// VGPR0-3: K = half*8 .. +7 ; VGPR4-7: K = 16 + half*8 .. +7
__device__ __forceinline__ v16bf load_A_frag(const bf16* __restrict__ base, int ld,
                                             int kbase, int lane) {
  const int m = lane & 15, hf = lane >> 4;
  const bf16* p = base + (size_t)m * ld + kbase + hf * 8;
  union { v16bf v; uint4 q[2]; } u;
  u.q[0] = *(const uint4*)(p);
  u.q[1] = *(const uint4*)(p + 16);
  return u.v;
}

// A fragment from the padded LDS tile (rows base_m..base_m+15 of 32)
__device__ __forceinline__ v16bf lds_A_frag(const bf16* tile, int base_m, int lane) {
  const int m = base_m + (lane & 15), hf = lane >> 4;
  const bf16* p = tile + m * AT_STRIDE + hf * 8;
  union { v16bf v; uint4 q[2]; } u;
  u.q[0] = *(const uint4*)(p);
  u.q[1] = *(const uint4*)(p + 16);
  return u.v;
}

// B fragment: 32x16 bf16, col N = lane&15, VGPR0-3: K = half*8 .. +7 ; VGPR4-7: K = 16 + half*8 .. +7.
// Source stored N-major rows with K contiguous: element(K,N) = base[N*ld + K].
__device__ __forceinline__ v16bf load_B_frag(const bf16* __restrict__ base, int ld,
                                             int kbase, int lane) {
  const int n = lane & 15, hf = lane >> 4;
  const bf16* p = base + (size_t)n * ld + kbase + hf * 8;
  union { v16bf v; uint4 q[2]; } u;
  u.q[0] = *(const uint4*)(p);
  u.q[1] = *(const uint4*)(p + 16);
  return u.v;
}

// ---------------- f32 -> bf16 conversion (4-wide) ----------------
__global__ void __launch_bounds__(256)
cvt_kernel(const float* __restrict__ src, bf16* __restrict__ dst, int n4) {
  int i = blockIdx.x * blockDim.x + threadIdx.x;
  if (i < n4) {
    float4 f = ((const float4*)src)[i];
    union { bf16 h[4]; uint2 u; } u;
    u.h[0] = (bf16)f.x; u.h[1] = (bf16)f.y;
    u.h[2] = (bf16)f.z; u.h[3] = (bf16)f.w;
    ((uint2*)dst)[i] = u.u;
  }
}

// ---- one pipeline stage: consume tile_cur + cb, prefetch kn into tile_next/nb
__device__ __forceinline__ void gemm_step(
    const bf16* __restrict__ Ab, const bf16* __restrict__ Wn,
    int lane, int row, int seg,
    bf16* tile_cur, bf16* tile_next, int kn,   // kn < 0: no prefetch
    v16bf cb[4], v16bf nb[4], v8f acc[2][4]) {
  wait_async0();
  block_barrier();                    // tile_cur populated; prev reads consumed
  if (kn >= 0) {
    async_copy16(Ab + (size_t)row * D_ + kn + seg * 8,
                 tile_next + row * AT_STRIDE + seg * 8);
#pragma unroll
    for (int t = 0; t < 4; ++t)
      nb[t] = load_B_frag(Wn + (size_t)(t * 16) * D_, D_, kn, lane);
  }
  v16bf a0 = lds_A_frag(tile_cur, 0, lane);
  v16bf a1 = lds_A_frag(tile_cur, 16, lane);
#pragma unroll
  for (int t = 0; t < 4; ++t) {
    acc[0][t] = wmma_bf16(a0, cb[t], acc[0][t]);
    acc[1][t] = wmma_bf16(a1, cb[t], acc[1][t]);
  }
}

// ---- shared GEMM core: block = 128 thr (4 waves), block tile 32M x 256N ----
// A rows streamed through double-buffered LDS via async copies (shared by all
// 4 waves); B fragments software-pipelined in two register sets (2x-unrolled
// K loop -> no register copies, no buffer-index selects).
__device__ __forceinline__ void gemm_core_32x256(
    const bf16* __restrict__ Ab,      // base at row gm0, ld = D_
    const bf16* __restrict__ Wn,      // base at weight row n0 (wave), ld = D_
    bf16 (*atile)[32 * AT_STRIDE],    // [2] ping-pong LDS buffers
    int tid, v8f acc[2][4]) {
  const int lane = tid & 31;
  const int row = tid >> 2;           // 0..31 : A tile row
  const int seg = tid & 3;            // 16B segment within 64B row chunk

  // preload chunk 0
  async_copy16(Ab + (size_t)row * D_ + seg * 8, &atile[0][row * AT_STRIDE + seg * 8]);
  v16bf fb0[4], fb1[4];
#pragma unroll
  for (int t = 0; t < 4; ++t)
    fb0[t] = load_B_frag(Wn + (size_t)(t * 16) * D_, D_, 0, lane);

  for (int kc = 0; kc < D_; kc += 64) {
    const int k1 = kc + 32;                          // always < D_
    const int k2 = (kc + 64 < D_) ? kc + 64 : -1;
    gemm_step(Ab, Wn, lane, row, seg, atile[0], atile[1], k1, fb0, fb1, acc);
    gemm_step(Ab, Wn, lane, row, seg, atile[1], atile[0], k2, fb1, fb0, acc);
  }
}

// ---------------- fused QKV projection ----------------
// gridDim = (BT/32, D/256, 3); block = 128
__global__ void __launch_bounds__(128)
qkv_kernel(const bf16* __restrict__ xb,
           const bf16* __restrict__ wqb, const bf16* __restrict__ wkb,
           const bf16* __restrict__ wvb,
           bf16* __restrict__ qb, bf16* __restrict__ kb, bf16* __restrict__ vtb,
           float* __restrict__ outk, float* __restrict__ outv) {
  __shared__ bf16 atile[2][32 * AT_STRIDE];
  const int tid  = threadIdx.x;
  const int lane = tid & 31;
  const int wave = tid >> 5;
  const int gm0  = blockIdx.x * 32;
  const int n0   = blockIdx.y * 256 + wave * 64;
  const int z    = blockIdx.z;
  const bf16* w = (z == 0) ? wqb : (z == 1) ? wkb : wvb;

  v8f zero = {0.f,0.f,0.f,0.f,0.f,0.f,0.f,0.f};
  v8f acc[2][4] = {{zero, zero, zero, zero}, {zero, zero, zero, zero}};

  gemm_core_32x256(xb + (size_t)gm0 * D_, w + (size_t)n0 * D_, atile, tid, acc);

  const int hf = lane >> 4, col = lane & 15;
#pragma unroll
  for (int mh = 0; mh < 2; ++mh) {
#pragma unroll
    for (int t = 0; t < 4; ++t) {
#pragma unroll
      for (int r = 0; r < 8; ++r) {
        int grow = gm0 + mh * 16 + r + 8 * hf;   // token over BT
        int gcol = n0 + t * 16 + col;            // feature over D
        float v  = acc[mh][t][r];
        int bb = grow >> 11;
        int tt = grow & (T_ - 1);
        int hh = gcol >> 6;
        int dd = gcol & (HD_ - 1);
        size_t idx = (((size_t)bb * H_ + hh) * T_ + tt) * HD_ + dd;
        if (z == 0) {
          qb[idx] = (bf16)(v * 0.125f);   // fold softmax 1/sqrt(HD) into Q
        } else if (z == 1) {
          outk[idx] = v;
          kb[idx]   = (bf16)v;
        } else {
          outv[idx] = v;
          vtb[(((size_t)bb * H_ + hh) * HD_ + dd) * T_ + tt] = (bf16)v;
        }
      }
    }
  }
}

// ---------------- flash attention ----------------
// gridDim = (T/16, B*H); block = 32 (one wave), tile = 16 queries x full head
__global__ void __launch_bounds__(32)
attn_kernel(const bf16* __restrict__ qb, const bf16* __restrict__ kb,
            const bf16* __restrict__ vtb, bf16* __restrict__ ab) {
  const int lane = threadIdx.x & 31;
  const int hf = lane >> 4, col = lane & 15;
  const int bh = blockIdx.y;
  const int bb = bh >> 4, hh = bh & 15;
  const int q0 = blockIdx.x * 16;

  const bf16* Qb = qb  + (size_t)bh * T_ * HD_;   // (t, d) ld=HD_ (pre-scaled)
  const bf16* Kb = kb  + (size_t)bh * T_ * HD_;   // (t, d) ld=HD_
  const bf16* Vt = vtb + (size_t)bh * HD_ * T_;   // (d, t) ld=T_

  __shared__ float pbuf[16 * 33];

  v16bf qa0 = load_A_frag(Qb + (size_t)q0 * HD_, HD_, 0, lane);
  v16bf qa1 = load_A_frag(Qb + (size_t)q0 * HD_, HD_, 32, lane);

  v8f zero = {0.f,0.f,0.f,0.f,0.f,0.f,0.f,0.f};
  v8f o[4] = {zero, zero, zero, zero};
  float m[8], l[8];
#pragma unroll
  for (int r = 0; r < 8; ++r) { m[r] = -3.0e38f; l[r] = 0.f; }

  const int kvend = q0 + 16;
  for (int kv0 = 0; kv0 < kvend; kv0 += 32) {
    if (kv0 + 32 < kvend) {
      __builtin_prefetch(Kb + (size_t)(kv0 + 32) * HD_ + lane * 16, 0, 0);
      __builtin_prefetch(Vt + (size_t)lane * T_ + kv0 + 32, 0, 0);
    }
    // S = Q K^T (Q pre-scaled)  (16 x 32 in two accumulators)
    v8f sa = zero, sb = zero;
    {
      v16bf bfr;
      bfr = load_B_frag(Kb + (size_t)kv0 * HD_, HD_, 0, lane);
      sa = wmma_bf16(qa0, bfr, sa);
      bfr = load_B_frag(Kb + (size_t)kv0 * HD_, HD_, 32, lane);
      sa = wmma_bf16(qa1, bfr, sa);
      bfr = load_B_frag(Kb + (size_t)(kv0 + 16) * HD_, HD_, 0, lane);
      sb = wmma_bf16(qa0, bfr, sb);
      bfr = load_B_frag(Kb + (size_t)(kv0 + 16) * HD_, HD_, 32, lane);
      sb = wmma_bf16(qa1, bfr, sb);
    }
    // issue V-tile fragment loads early: independent of the softmax VALU phase
    v16bf bv[4];
#pragma unroll
    for (int a = 0; a < 4; ++a)
      bv[a] = load_B_frag(Vt + (size_t)(a * 16) * T_, T_, kv0, lane);

    if (kv0 + 31 > q0) {            // causal mask only on boundary chunk
      int colA = kv0 + col, colB = colA + 16;
#pragma unroll
      for (int r = 0; r < 8; ++r) {
        int row = q0 + r + 8 * hf;
        if (colA > row) sa[r] = -3.0e38f;
        if (colB > row) sb[r] = -3.0e38f;
      }
    }

    // online softmax (per-row butterfly across 16-lane halves)
#pragma unroll
    for (int r = 0; r < 8; ++r) {
      float mx = fmaxf(sa[r], sb[r]);
      mx = fmaxf(mx, __shfl_xor(mx, 1, 32));
      mx = fmaxf(mx, __shfl_xor(mx, 2, 32));
      mx = fmaxf(mx, __shfl_xor(mx, 4, 32));
      mx = fmaxf(mx, __shfl_xor(mx, 8, 32));
      float mn = fmaxf(m[r], mx);
      float corr = __expf(m[r] - mn);
      m[r] = mn;
      float pa = __expf(sa[r] - mn);
      float pb = __expf(sb[r] - mn);
      float rs = pa + pb;
      rs += __shfl_xor(rs, 1, 32);
      rs += __shfl_xor(rs, 2, 32);
      rs += __shfl_xor(rs, 4, 32);
      rs += __shfl_xor(rs, 8, 32);
      l[r] = l[r] * corr + rs;
#pragma unroll
      for (int a = 0; a < 4; ++a) o[a][r] *= corr;
      sa[r] = pa; sb[r] = pb;
    }

    // C-layout P -> LDS -> A-fragment layout (bf16)
#pragma unroll
    for (int r = 0; r < 8; ++r) {
      pbuf[(r + 8 * hf) * 33 + col]      = sa[r];
      pbuf[(r + 8 * hf) * 33 + col + 16] = sb[r];
    }
    __syncthreads();
    v16bf pf;
    {
      const int mrow = lane & 15;
#pragma unroll
      for (int j = 0; j < 8; ++j) {
        int k0 = ((j & 4) ? 16 : 0) + hf * 8 + (j & 3) * 2;
        pf[2 * j]     = (bf16)pbuf[mrow * 33 + k0];
        pf[2 * j + 1] = (bf16)pbuf[mrow * 33 + k0 + 1];
      }
    }
    __syncthreads();

    // O += P * V
#pragma unroll
    for (int a = 0; a < 4; ++a)
      o[a] = wmma_bf16(pf, bv[a], o[a]);
  }

  // epilogue: O / l  -> attn_out bf16 in (b, t, D) layout
#pragma unroll
  for (int a = 0; a < 4; ++a) {
#pragma unroll
    for (int r = 0; r < 8; ++r) {
      float v = o[a][r] / l[r];
      int row  = q0 + r + 8 * hf;
      int dcol = a * 16 + col;
      ab[((size_t)bb * T_ + row) * D_ + hh * HD_ + dcol] = (bf16)v;
    }
  }
}

// ---------------- output projection ----------------
// gridDim = (BT/32, D/256); block = 128
__global__ void __launch_bounds__(128)
oproj_kernel(const bf16* __restrict__ ab, const bf16* __restrict__ wob,
             float* __restrict__ out) {
  __shared__ bf16 atile[2][32 * AT_STRIDE];
  const int tid  = threadIdx.x;
  const int lane = tid & 31;
  const int wave = tid >> 5;
  const int gm0  = blockIdx.x * 32;
  const int n0   = blockIdx.y * 256 + wave * 64;

  v8f zero = {0.f,0.f,0.f,0.f,0.f,0.f,0.f,0.f};
  v8f acc[2][4] = {{zero, zero, zero, zero}, {zero, zero, zero, zero}};

  gemm_core_32x256(ab + (size_t)gm0 * D_, wob + (size_t)n0 * D_, atile, tid, acc);

  const int hf = lane >> 4, col = lane & 15;
#pragma unroll
  for (int mh = 0; mh < 2; ++mh) {
#pragma unroll
    for (int t = 0; t < 4; ++t) {
#pragma unroll
      for (int r = 0; r < 8; ++r) {
        int grow = gm0 + mh * 16 + r + 8 * hf;
        int gcol = n0 + t * 16 + col;
        out[(size_t)grow * D_ + gcol] = acc[mh][t][r];
      }
    }
  }
}

extern "C" void kernel_launch(void* const* d_in, const int* in_sizes, int n_in,
                              void* d_out, int out_size, void* d_ws, size_t ws_size,
                              hipStream_t stream) {
  const float* x  = (const float*)d_in[0];
  const float* wq = (const float*)d_in[1];
  const float* wk = (const float*)d_in[2];
  const float* wv = (const float*)d_in[3];
  const float* wo = (const float*)d_in[4];
  float* out  = (float*)d_out;
  float* outk = out + OUTK_ELEM_OFF;
  float* outv = out + OUTV_ELEM_OFF;

  char* ws = (char*)d_ws;
  bf16* xb  = (bf16*)(ws + XB_OFF);
  bf16* wqb = (bf16*)(ws + WQ_OFF);
  bf16* wkb = (bf16*)(ws + WK_OFF);
  bf16* wvb = (bf16*)(ws + WV_OFF);
  bf16* wob = (bf16*)(ws + WO_OFF);
  bf16* qb  = (bf16*)(ws + QB_OFF);
  bf16* kb  = (bf16*)(ws + KB_OFF);
  bf16* vtb = (bf16*)(ws + VT_OFF);
  bf16* abf = (bf16*)(ws + AB_OFF);

  // f32 -> bf16 staging
  {
    int n4x = (BT_ * D_) / 4;
    cvt_kernel<<<n4x / 256, 256, 0, stream>>>(x, xb, n4x);
    int n4w = (D_ * D_) / 4;
    cvt_kernel<<<n4w / 256, 256, 0, stream>>>(wq, wqb, n4w);
    cvt_kernel<<<n4w / 256, 256, 0, stream>>>(wk, wkb, n4w);
    cvt_kernel<<<n4w / 256, 256, 0, stream>>>(wv, wvb, n4w);
    cvt_kernel<<<n4w / 256, 256, 0, stream>>>(wo, wob, n4w);
  }

  // QKV projections (also emits k, v f32 outputs)
  {
    dim3 grid(BT_ / 32, D_ / 256, 3);
    qkv_kernel<<<grid, 128, 0, stream>>>(xb, wqb, wkb, wvb, qb, kb, vtb, outk, outv);
  }

  // causal flash attention
  {
    dim3 grid(T_ / 16, B_ * H_);
    attn_kernel<<<grid, 32, 0, stream>>>(qb, kb, vtb, abf);
  }

  // output projection
  {
    dim3 grid(BT_ / 32, D_ / 256);
    oproj_kernel<<<grid, 128, 0, stream>>>(abf, wob, out);
  }
}
